// BidirectionalLaCTSwiGLU_21285857919693
// MI455X (gfx1250) — matrix operand order repaired and checked by reading, not compiled, not run on hardware
//
#include <hip/hip_runtime.h>
#include <hip/hip_bf16.h>

typedef _Float16 half_t;
typedef __attribute__((ext_vector_type(16))) _Float16 v16h;
typedef __attribute__((ext_vector_type(8)))  _Float16 v8h;
typedef __attribute__((ext_vector_type(8)))  float    v8f;

#define DIMX 768
#define HDX  64
#define NHX  12
#define NBX  8
#define LLX  2048
#define BHX  (NBX * NHX)   // 96

// ---------------------------------------------------------------------------
// WMMA helpers (wave32, 16x16x32 f16 -> f32)
// ---------------------------------------------------------------------------
__device__ __forceinline__ v8f wmma_f16(v16h a, v16h b, v8f c) {
  return __builtin_amdgcn_wmma_f32_16x16x32_f16(false, a, false, b, (short)0, c,
                                                false, false);
}

// A tile 16x32, row-major source: A[m][k] = S[m*ld + k]
__device__ __forceinline__ v16h ld_a_row(const half_t* S, int ld) {
  const int lane = threadIdx.x & 31;
  const int m = lane & 15;
  const int kb = (lane >> 4) << 3;  // 0 or 8
  const half_t* p = S + m * ld + kb;
  v8h lo = *(const v8h*)p;
  v8h hi = *(const v8h*)(p + 16);
  v16h r;
#pragma unroll
  for (int i = 0; i < 8; ++i) { r[i] = lo[i]; r[8 + i] = hi[i]; }
  return r;
}

// A tile 16x32, transposed source: A[m][k] = S[k*ld + m]
__device__ __forceinline__ v16h ld_a_T(const half_t* S, int ld) {
  const int lane = threadIdx.x & 31;
  const int m = lane & 15;
  const int kb = (lane >> 4) << 3;
  v16h r;
#pragma unroll
  for (int i = 0; i < 8; ++i) {
    r[i]     = S[(kb + i) * ld + m];
    r[8 + i] = S[(kb + 16 + i) * ld + m];
  }
  return r;
}

// B tile 32x16 from transposed source: B[k][n] = S[n*ld + k]
__device__ __forceinline__ v16h ld_b_T(const half_t* S, int ld) {
  const int lane = threadIdx.x & 31;
  const int n = lane & 15;
  const int kb = (lane >> 4) << 3;  // 0 or 8
  const half_t* p = S + n * ld + kb;
  v8h lo = *(const v8h*)p;
  v8h hi = *(const v8h*)(p + 16);
  v16h r;
#pragma unroll
  for (int i = 0; i < 8; ++i) { r[i] = lo[i]; r[8 + i] = hi[i]; }
  return r;
}

// B tile 32x16, row-major source: B[k][n] = S[k*ld + n]
__device__ __forceinline__ v16h ld_b_row(const half_t* S, int ld) {
  const int lane = threadIdx.x & 31;
  const int n = lane & 15;
  const int kb = (lane >> 4) << 3;
  v16h r;
#pragma unroll
  for (int i = 0; i < 16; ++i) r[i] = S[(kb + i + ((i >> 3) << 3)) * ld + n];
  return r;
}

__device__ __forceinline__ void st_c_f32(float* C, int ld, v8f c) {
  const int lane = threadIdx.x & 31;
  const int n = lane & 15;
  const int mb = (lane >> 4) << 3;
#pragma unroll
  for (int e = 0; e < 8; ++e) C[(mb + e) * ld + n] = c[e];
}

__device__ __forceinline__ void st_c_f16(half_t* C, int ld, v8f c) {
  const int lane = threadIdx.x & 31;
  const int n = lane & 15;
  const int mb = (lane >> 4) << 3;
#pragma unroll
  for (int e = 0; e < 8; ++e) C[(mb + e) * ld + n] = (half_t)c[e];
}

// ---------------------------------------------------------------------------
// CDNA5 async global->LDS staging (ASYNCcnt path; bypasses VGPRs).
// Generic pointer low 32 bits == LDS byte offset (flat LDS aperture).
// ---------------------------------------------------------------------------
__device__ __forceinline__ void async_g2l_b128(void* lds_ptr, const void* gptr) {
  unsigned loff = (unsigned)(size_t)lds_ptr;
  asm volatile("global_load_async_to_lds_b128 %0, %1, off"
               :
               : "v"(loff), "v"(gptr)
               : "memory");
}
__device__ __forceinline__ void wait_async0() {
  asm volatile("s_wait_asynccnt 0x0" ::: "memory");
}

// ---------------------------------------------------------------------------
// 1) f32 -> f16 cast
// ---------------------------------------------------------------------------
__global__ void cast_f32_f16(const float* __restrict__ s, half_t* __restrict__ d,
                             int n) {
  int i = blockIdx.x * 256 + threadIdx.x;
  if (i < n) d[i] = (half_t)s[i];
}

// ---------------------------------------------------------------------------
// 2) per-token per-head learning rates: softplus(x @ lr_w.T + b + c0)
// ---------------------------------------------------------------------------
__global__ void lr_kernel(const float* __restrict__ x, const float* __restrict__ lw,
                          const float* __restrict__ lb, float* __restrict__ lr0,
                          float* __restrict__ lr1, float* __restrict__ lr2) {
  int idx = blockIdx.x * 256 + threadIdx.x;  // 16384*36
  int row = idx / 36;
  int j = idx - row * 36;
  const float* xp = x + (size_t)row * DIMX;
  const float* wp = lw + (size_t)j * DIMX;
  float s = 0.f;
  for (int d = 0; d < DIMX; ++d) s += xp[d] * wp[d];
  float v = s + lb[j] + (-4.600166f);  // inv_softplus(0.01)
  float sp = (v > 20.f) ? v : log1pf(__expf(v));
  int kind = j / NHX, h = j - kind * NHX;
  int b = row >> 11, l = row & 2047;
  float* dst = (kind == 0) ? lr0 : (kind == 1) ? lr1 : lr2;
  dst[((size_t)(b * NHX + h)) * LLX + l] = sp;
}

// ---------------------------------------------------------------------------
// 3) qkv projection GEMM (WMMA), wave computes a 16x64 strip: one A tile is
//    reused across 4 B tiles per K-step. Epilogue scatters into head-major
//    q/k/v f16.  C[16384,2304] = Xh[16384,768] @ qkv_w^T
// ---------------------------------------------------------------------------
__global__ void __launch_bounds__(256) gemm_qkv(const half_t* __restrict__ A,
                                                const half_t* __restrict__ Wt,
                                                half_t* __restrict__ qh,
                                                half_t* __restrict__ kh,
                                                half_t* __restrict__ vh) {
  const int wave = threadIdx.x >> 5;
  const int job = blockIdx.x * 8 + wave;  // 1024 * 36 strips
  const int tg = job % 36;                // group of 4 N-tiles
  const int tm = job / 36;
  v8f acc[4];
#pragma unroll
  for (int j = 0; j < 4; ++j) acc[j] = {};
  const half_t* Ap = A + (size_t)tm * 16 * DIMX;
  const half_t* Bp = Wt + (size_t)tg * 64 * DIMX;
#pragma unroll 2
  for (int k0 = 0; k0 < DIMX; k0 += 32) {
    __builtin_prefetch(Ap + k0 + 128, 0, 1);
    __builtin_prefetch(Bp + k0 + 128, 0, 1);
    v16h a = ld_a_row(Ap + k0, DIMX);
#pragma unroll
    for (int j = 0; j < 4; ++j) {
      v16h b = ld_b_T(Bp + (size_t)j * 16 * DIMX + k0, DIMX);
      acc[j] = wmma_f16(a, b, acc[j]);
    }
  }
  const int lane = threadIdx.x & 31;
  const int mb = (lane >> 4) << 3;
#pragma unroll
  for (int j = 0; j < 4; ++j) {
    const int n = (tg * 4 + j) * 16 + (lane & 15);
    const int part = n / DIMX;
    const int rem = n - part * DIMX;
    const int h = rem >> 6, hd = rem & 63;
    half_t* dst = (part == 0) ? qh : (part == 1) ? kh : vh;
#pragma unroll
    for (int e = 0; e < 8; ++e) {
      int r = tm * 16 + mb + e;
      int b_ = r >> 11, l = r & 2047;
      dst[(((size_t)(b_ * NHX + h)) * LLX + l) * HDX + hd] = (half_t)acc[j][e];
    }
  }
}

// ---------------------------------------------------------------------------
// 4) q/k L2 normalization over head_dim, in place
// ---------------------------------------------------------------------------
__global__ void qknorm(half_t* __restrict__ qh, half_t* __restrict__ kh) {
  size_t i = (size_t)blockIdx.x * 256 + threadIdx.x;  // BHX*LLX rows
  half_t* q = qh + i * HDX;
  half_t* k = kh + i * HDX;
  float sq = 0.f, sk = 0.f;
  for (int d = 0; d < HDX; ++d) {
    float a = (float)q[d]; sq += a * a;
    float b = (float)k[d]; sk += b * b;
  }
  float cq = 1.f / (sqrtf(sq) + 1e-5f);
  float ck = 1.f / (sqrtf(sk) + 1e-5f);
  for (int d = 0; d < HDX; ++d) {
    q[d] = (half_t)((float)q[d] * cq);
    k[d] = (half_t)((float)k[d] * ck);
  }
}

// ---------------------------------------------------------------------------
// 5) fast-weight gradient accumulation per head (WMMA everywhere)
//    dw0 = dgate_pre @ (k*lr0); dw1 = v^T @ (hidden^T*lr1); dw2 = dhid_pre @ (k*lr2)
//    k/v chunks staged with GLOBAL_LOAD_ASYNC_TO_LDS_B128.
// ---------------------------------------------------------------------------
__global__ void __launch_bounds__(256) dw_kernel(
    const half_t* __restrict__ kh, const half_t* __restrict__ vh,
    const float* __restrict__ w0g, const float* __restrict__ w1g,
    const float* __restrict__ w2g, const float* __restrict__ lr0p,
    const float* __restrict__ lr1p, const float* __restrict__ lr2p,
    float* __restrict__ dwb) {
  __shared__ __attribute__((aligned(16))) half_t w0s[64 * 64];
  __shared__ __attribute__((aligned(16))) half_t w1s[64 * 64];
  __shared__ __attribute__((aligned(16))) half_t w2s[64 * 64];
  __shared__ __attribute__((aligned(16))) half_t ks[32 * 64];
  __shared__ __attribute__((aligned(16))) half_t vs[32 * 64];
  __shared__ __attribute__((aligned(16))) half_t klr0T[64 * 40];
  __shared__ __attribute__((aligned(16))) half_t klr2T[64 * 40];
  __shared__ __attribute__((aligned(16))) half_t hids[64 * 40];
  __shared__ __attribute__((aligned(16))) half_t dgs[64 * 40];
  __shared__ __attribute__((aligned(16))) half_t dhs[64 * 40];
  __shared__ float lr0s[32], lr1s[32], lr2s[32];

  const int t = threadIdx.x;
  const int bh = blockIdx.x;
  const int h = bh % NHX;
  const int wave = t >> 5;
  const int lane = t & 31;
  const int n_l = lane & 15;
  const int mb = (lane >> 4) << 3;

  const float* s0 = w0g + (size_t)h * 4096;
  const float* s1 = w1g + (size_t)h * 4096;
  const float* s2 = w2g + (size_t)h * 4096;
  for (int i = t; i < 4096; i += 256) {
    w0s[i] = (half_t)s0[i];
    w1s[i] = (half_t)s1[i];
    w2s[i] = (half_t)s2[i];
  }
  __syncthreads();

  const int fmt = wave >> 1, fnt = wave & 1;  // forward tile ownership
  const int t0 = wave * 2;                    // dw tile ownership

  v8f a0[2], a1[2], a2[2];
#pragma unroll
  for (int j = 0; j < 2; ++j) { a0[j] = {}; a1[j] = {}; a2[j] = {}; }

  const half_t* kbase = kh + (size_t)bh * LLX * HDX;
  const half_t* vbase = vh + (size_t)bh * LLX * HDX;

  for (int c = 0; c < LLX / 32; ++c) {
    const int l0 = c * 32;
    // async stage: 256 lanes x 16B == one 32x64 f16 chunk each
    async_g2l_b128(&ks[t * 8], kbase + (size_t)l0 * 64 + t * 8);
    async_g2l_b128(&vs[t * 8], vbase + (size_t)l0 * 64 + t * 8);
    if (t < 32)      lr0s[t]      = lr0p[(size_t)bh * LLX + l0 + t];
    else if (t < 64) lr1s[t - 32] = lr1p[(size_t)bh * LLX + l0 + t - 32];
    else if (t < 96) lr2s[t - 64] = lr2p[(size_t)bh * LLX + l0 + t - 64];
    wait_async0();
    __syncthreads();

    // k*lr0, k*lr2 transposed to [hd][l]
    for (int i = t; i < 2048; i += 256) {
      int hd = i >> 5, l = i & 31;
      float kv = (float)ks[l * 64 + hd];
      klr0T[hd * 40 + l] = (half_t)(kv * lr0s[l]);
      klr2T[hd * 40 + l] = (half_t)(kv * lr2s[l]);
    }

    // forward: gate_pre = w0@k^T, hid_pre = w2@k^T, dhidden = w1^T@v^T
    v8f gp = {}, hp = {}, dhv = {};
#pragma unroll
    for (int kk = 0; kk < 2; ++kk) {
      v16h bk  = ld_b_T(ks + fnt * 16 * 64 + kk * 32, 64);
      v16h bv  = ld_b_T(vs + fnt * 16 * 64 + kk * 32, 64);
      v16h aw0 = ld_a_row(w0s + fmt * 16 * 64 + kk * 32, 64);
      v16h aw2 = ld_a_row(w2s + fmt * 16 * 64 + kk * 32, 64);
      v16h aw1 = ld_a_T(w1s + kk * 32 * 64 + fmt * 16, 64);
      gp  = wmma_f16(aw0, bk, gp);
      hp  = wmma_f16(aw2, bk, hp);
      dhv = wmma_f16(aw1, bv, dhv);
    }
#pragma unroll
    for (int e = 0; e < 8; ++e) {
      float g = gp[e], hh = hp[e], d = dhv[e];
      float sg = 1.f / (1.f + __expf(-g));
      float silu = g * sg;
      float hidden = silu * hh;
      float dhp = d * silu;
      float dgp = d * hh * sg * (1.f + g * (1.f - sg));
      int row = fmt * 16 + mb + e;
      int col = fnt * 16 + n_l;
      hids[row * 40 + col] = (half_t)(hidden * lr1s[col]);
      dgs[row * 40 + col]  = (half_t)dgp;
      dhs[row * 40 + col]  = (half_t)dhp;
    }
    __syncthreads();

    // accumulate dw (K = this 32-token chunk)
#pragma unroll
    for (int j = 0; j < 2; ++j) {
      int tt = t0 + j;
      int mt = tt >> 2, nt = tt & 3;
      v16h aA  = ld_a_row(dgs + mt * 16 * 40, 40);
      v16h bB  = ld_b_T(klr0T + nt * 16 * 40, 40);
      a0[j] = wmma_f16(aA, bB, a0[j]);
      v16h aV  = ld_a_T(vs + mt * 16, 64);
      v16h bH  = ld_b_T(hids + nt * 16 * 40, 40);
      a1[j] = wmma_f16(aV, bH, a1[j]);
      v16h aD  = ld_a_row(dhs + mt * 16 * 40, 40);
      v16h bK2 = ld_b_T(klr2T + nt * 16 * 40, 40);
      a2[j] = wmma_f16(aD, bK2, a2[j]);
    }
    __syncthreads();
  }

#pragma unroll
  for (int j = 0; j < 2; ++j) {
    int tt = t0 + j;
    int mt = tt >> 2, nt = tt & 3;
    size_t base = (size_t)bh * 4096 + mt * 16 * 64 + nt * 16;
    st_c_f32(dwb + 0 * (size_t)BHX * 4096 + base, 64, a0[j]);
    st_c_f32(dwb + 1 * (size_t)BHX * 4096 + base, 64, a1[j]);
    st_c_f32(dwb + 2 * (size_t)BHX * 4096 + base, 64, a2[j]);
  }
}

// ---------------------------------------------------------------------------
// 6) Newton-Schulz (5 iters) on 64x64 in f16 WMMA + renorm; w = ns5(dw)+dw
// ---------------------------------------------------------------------------
__global__ void __launch_bounds__(256) ns5_kernel(const float* __restrict__ dwb,
                                                  const float* __restrict__ w0g,
                                                  const float* __restrict__ w1g,
                                                  const float* __restrict__ w2g,
                                                  half_t* __restrict__ wnew) {
  __shared__ __attribute__((aligned(16))) half_t X[64 * 72];
  __shared__ __attribute__((aligned(16))) half_t Am[64 * 72];
  __shared__ float wf[64 * 64];
  __shared__ float red[256];
  __shared__ float rno[64], rnn[64];

  const int t = threadIdx.x;
  const int mat = blockIdx.x % 3;
  const int bh = blockIdx.x / 3;
  const int h = bh % NHX;
  const int wave = t >> 5;
  const int lane = t & 31;
  const int n_l = lane & 15;
  const int mb = (lane >> 4) << 3;

  const float* G = dwb + ((size_t)mat * BHX + bh) * 4096;
  const float* worig =
      ((mat == 0) ? w0g : (mat == 1) ? w1g : w2g) + (size_t)h * 4096;

  if (t < 64) {
    float s = 0.f;
    for (int c2 = 0; c2 < 64; ++c2) { float v = worig[t * 64 + c2]; s += v * v; }
    rno[t] = sqrtf(s);
  }
  float s = 0.f;
  for (int i = t; i < 4096; i += 256) { float v = G[i]; s += v * v; }
  red[t] = s;
  __syncthreads();
  for (int st = 128; st > 0; st >>= 1) {
    if (t < st) red[t] += red[t + st];
    __syncthreads();
  }
  float inv = 1.f / (sqrtf(red[0]) + 1e-7f);
  __syncthreads();
  for (int i = t; i < 4096; i += 256)
    X[(i >> 6) * 72 + (i & 63)] = (half_t)(G[i] * inv);
  __syncthreads();

  const float NSa[5] = {4.0848f, 3.9505f, 3.7418f, 2.8769f, 2.8366f};
  const float NSb[5] = {-6.8946f, -6.3029f, -5.5913f, -3.1427f, -3.0525f};
  const float NSc[5] = {2.927f, 2.6377f, 2.3037f, 1.2046f, 1.2012f};

  const int t0 = wave * 2;

  for (int it = 0; it < 5; ++it) {
    // A = X X^T
#pragma unroll
    for (int j = 0; j < 2; ++j) {
      int tt = t0 + j, mt = tt >> 2, nt = tt & 3;
      v8f acc = {};
#pragma unroll
      for (int kk = 0; kk < 2; ++kk) {
        v16h a = ld_a_row(X + mt * 16 * 72 + kk * 32, 72);
        v16h b = ld_b_T(X + nt * 16 * 72 + kk * 32, 72);
        acc = wmma_f16(a, b, acc);
      }
      st_c_f16(Am + mt * 16 * 72 + nt * 16, 72, acc);
    }
    __syncthreads();
    // A2 = A @ A  (A symmetric -> contiguous B loads)
    v8f A2[2];
#pragma unroll
    for (int j = 0; j < 2; ++j) {
      int tt = t0 + j, mt = tt >> 2, nt = tt & 3;
      v8f acc = {};
#pragma unroll
      for (int kk = 0; kk < 2; ++kk) {
        v16h a = ld_a_row(Am + mt * 16 * 72 + kk * 32, 72);
        v16h b = ld_b_T(Am + nt * 16 * 72 + kk * 32, 72);
        acc = wmma_f16(a, b, acc);
      }
      A2[j] = acc;
    }
    __syncthreads();
    // Bm = b*A + c*A2 (overwrite own tiles of Am)
#pragma unroll
    for (int j = 0; j < 2; ++j) {
      int tt = t0 + j, mt = tt >> 2, nt = tt & 3;
#pragma unroll
      for (int e = 0; e < 8; ++e) {
        int row = mt * 16 + mb + e, col = nt * 16 + n_l;
        float Ae = (float)Am[row * 72 + col];
        Am[row * 72 + col] = (half_t)(NSb[it] * Ae + NSc[it] * A2[j][e]);
      }
    }
    __syncthreads();
    // Xn = a*X + Bm @ X
    v8f Xn[2];
#pragma unroll
    for (int j = 0; j < 2; ++j) {
      int tt = t0 + j, mt = tt >> 2, nt = tt & 3;
      v8f acc = {};
#pragma unroll
      for (int kk = 0; kk < 2; ++kk) {
        v16h a = ld_a_row(Am + mt * 16 * 72 + kk * 32, 72);
        v16h b = ld_b_row(X + kk * 32 * 72 + nt * 16, 72);
        acc = wmma_f16(a, b, acc);
      }
#pragma unroll
      for (int e = 0; e < 8; ++e) {
        int row = mt * 16 + mb + e, col = nt * 16 + n_l;
        acc[e] += NSa[it] * (float)X[row * 72 + col];
      }
      Xn[j] = acc;
    }
    __syncthreads();
#pragma unroll
    for (int j = 0; j < 2; ++j) {
      int tt = t0 + j, mt = tt >> 2, nt = tt & 3;
      st_c_f16(X + mt * 16 * 72 + nt * 16, 72, Xn[j]);
    }
    __syncthreads();
  }

  // w = X + dw ; row renorm to original row norms
  for (int i = t; i < 4096; i += 256)
    wf[i] = (float)X[(i >> 6) * 72 + (i & 63)] + G[i];
  __syncthreads();
  if (t < 64) {
    float s2 = 0.f;
    for (int c2 = 0; c2 < 64; ++c2) { float v = wf[t * 64 + c2]; s2 += v * v; }
    rnn[t] = sqrtf(s2);
  }
  __syncthreads();
  half_t* out = wnew + ((size_t)mat * BHX + bh) * 4096;
  for (int i = t; i < 4096; i += 256) {
    int r = i >> 6;
    out[i] = (half_t)(wf[i] / (rnn[r] + 1e-5f) * rno[r]);
  }
}

// ---------------------------------------------------------------------------
// 7) output pass per head: o = w1n @ (silu(w0n@q^T) * (w2n@q^T)), RMSNorm
//    q chunks staged with GLOBAL_LOAD_ASYNC_TO_LDS_B128.
// ---------------------------------------------------------------------------
__global__ void __launch_bounds__(256) outhead_kernel(
    const half_t* __restrict__ qh, const half_t* __restrict__ wnew,
    const float* __restrict__ onormw, half_t* __restrict__ osb) {
  __shared__ __attribute__((aligned(16))) half_t w0s[64 * 64];
  __shared__ __attribute__((aligned(16))) half_t w1s[64 * 64];
  __shared__ __attribute__((aligned(16))) half_t w2s[64 * 64];
  __shared__ __attribute__((aligned(16))) half_t qs[32 * 64];
  __shared__ __attribute__((aligned(16))) half_t ghs[64 * 40];
  __shared__ float oL[64 * 32];
  __shared__ float red[256];
  __shared__ float rmsv[32];
  __shared__ float onw[64];

  const int t = threadIdx.x;
  const int bh = blockIdx.x;
  const int b = bh / NHX, h = bh % NHX;
  const int wave = t >> 5;
  const int lane = t & 31;
  const int n_l = lane & 15;
  const int mb = (lane >> 4) << 3;

  for (int i = t; i < 4096; i += 256) {
    w0s[i] = wnew[(size_t)(0 * BHX + bh) * 4096 + i];
    w1s[i] = wnew[(size_t)(1 * BHX + bh) * 4096 + i];
    w2s[i] = wnew[(size_t)(2 * BHX + bh) * 4096 + i];
  }
  if (t < 64) onw[t] = onormw[t];
  __syncthreads();

  const half_t* qbase = qh + (size_t)bh * LLX * HDX;
  const int fmt = wave >> 1, fnt = wave & 1;

  for (int c = 0; c < LLX / 32; ++c) {
    const int l0 = c * 32;
    async_g2l_b128(&qs[t * 8], qbase + (size_t)l0 * 64 + t * 8);
    wait_async0();
    __syncthreads();

    v8f gp = {}, hp = {};
#pragma unroll
    for (int kk = 0; kk < 2; ++kk) {
      v16h bq  = ld_b_T(qs + fnt * 16 * 64 + kk * 32, 64);
      v16h aw0 = ld_a_row(w0s + fmt * 16 * 64 + kk * 32, 64);
      v16h aw2 = ld_a_row(w2s + fmt * 16 * 64 + kk * 32, 64);
      gp = wmma_f16(aw0, bq, gp);
      hp = wmma_f16(aw2, bq, hp);
    }
#pragma unroll
    for (int e = 0; e < 8; ++e) {
      float g = gp[e];
      float sg = 1.f / (1.f + __expf(-g));
      int row = fmt * 16 + mb + e, col = fnt * 16 + n_l;
      ghs[row * 40 + col] = (half_t)(g * sg * hp[e]);
    }
    __syncthreads();

    v8f o = {};
#pragma unroll
    for (int kk = 0; kk < 2; ++kk) {
      v16h a = ld_a_row(w1s + fmt * 16 * 64 + kk * 32, 64);
      v16h b = ld_b_row(ghs + kk * 32 * 40 + fnt * 16, 40);
      o = wmma_f16(a, b, o);
    }
    st_c_f32(oL + fmt * 16 * 32 + fnt * 16, 32, o);
    __syncthreads();

    // column RMS over 64 head dims
    {
      int col = t & 31, seg = t >> 5;
      float s = 0.f;
      for (int r = seg * 8; r < seg * 8 + 8; ++r) {
        float v = oL[r * 32 + col];
        s += v * v;
      }
      red[t] = s;
    }
    __syncthreads();
    if (t < 32) {
      float s = 0.f;
      for (int seg = 0; seg < 8; ++seg) s += red[seg * 32 + t];
      rmsv[t] = sqrtf(s * (1.f / 64.f) + 1e-5f);
    }
    __syncthreads();
    for (int i = t; i < 2048; i += 256) {
      int hd = i >> 5, l = i & 31;
      float v = oL[hd * 32 + l] / rmsv[l] * onw[hd];
      osb[((size_t)b * LLX + l0 + l) * DIMX + h * HDX + hd] = (half_t)v;
    }
    __syncthreads();
  }
}

// ---------------------------------------------------------------------------
// 8) output projection GEMM: out[16384,768] = o_s @ o_proj_w^T (f32 out),
//    wave computes a 16x64 strip (A tile reused across 4 B tiles).
// ---------------------------------------------------------------------------
__global__ void __launch_bounds__(256) gemm_oproj(const half_t* __restrict__ A,
                                                  const half_t* __restrict__ Wt,
                                                  float* __restrict__ out) {
  const int wave = threadIdx.x >> 5;
  const int job = blockIdx.x * 8 + wave;  // 1024 * 12 strips
  const int tg = job % 12;
  const int tm = job / 12;
  v8f acc[4];
#pragma unroll
  for (int j = 0; j < 4; ++j) acc[j] = {};
  const half_t* Ap = A + (size_t)tm * 16 * DIMX;
  const half_t* Bp = Wt + (size_t)tg * 64 * DIMX;
#pragma unroll 2
  for (int k0 = 0; k0 < DIMX; k0 += 32) {
    __builtin_prefetch(Ap + k0 + 128, 0, 1);
    __builtin_prefetch(Bp + k0 + 128, 0, 1);
    v16h a = ld_a_row(Ap + k0, DIMX);
#pragma unroll
    for (int j = 0; j < 4; ++j) {
      v16h b = ld_b_T(Bp + (size_t)j * 16 * DIMX + k0, DIMX);
      acc[j] = wmma_f16(a, b, acc[j]);
    }
  }
#pragma unroll
  for (int j = 0; j < 4; ++j)
    st_c_f32(out + (size_t)tm * 16 * DIMX + (tg * 4 + j) * 16, DIMX, acc[j]);
}

// ---------------------------------------------------------------------------
extern "C" void kernel_launch(void* const* d_in, const int* in_sizes, int n_in,
                              void* d_out, int out_size, void* d_ws,
                              size_t ws_size, hipStream_t stream) {
  (void)in_sizes; (void)n_in; (void)out_size; (void)ws_size;
  const float* x      = (const float*)d_in[0];
  const float* qkvw   = (const float*)d_in[1];
  const float* lrw    = (const float*)d_in[2];
  const float* lrb    = (const float*)d_in[3];
  const float* w0     = (const float*)d_in[4];
  const float* w1     = (const float*)d_in[5];
  const float* w2     = (const float*)d_in[6];
  const float* onormw = (const float*)d_in[7];
  const float* oprojw = (const float*)d_in[8];
  float* out = (float*)d_out;

  char* W = (char*)d_ws;
  size_t off = 0;
  auto take = [&](size_t bytes) -> char* {
    char* p = W + off;
    off += (bytes + 255) & ~(size_t)255;
    return p;
  };
  half_t* x_h  = (half_t*)take((size_t)16384 * 768 * 2);
  half_t* qw_h = (half_t*)take((size_t)2304 * 768 * 2);
  half_t* ow_h = (half_t*)take((size_t)768 * 768 * 2);
  half_t* q_h  = (half_t*)take((size_t)96 * 2048 * 64 * 2);
  half_t* k_h  = (half_t*)take((size_t)96 * 2048 * 64 * 2);
  half_t* v_h  = (half_t*)take((size_t)96 * 2048 * 64 * 2);
  float*  lr0p = (float*)take((size_t)96 * 2048 * 4);
  float*  lr1p = (float*)take((size_t)96 * 2048 * 4);
  float*  lr2p = (float*)take((size_t)96 * 2048 * 4);
  float*  dwb  = (float*)take((size_t)3 * 96 * 4096 * 4);
  half_t* wnew = (half_t*)take((size_t)3 * 96 * 4096 * 2);
  half_t* o_s  = (half_t*)take((size_t)16384 * 768 * 2);

  cast_f32_f16<<<(16384 * 768) / 256, 256, 0, stream>>>(x, x_h, 16384 * 768);
  cast_f32_f16<<<(2304 * 768) / 256, 256, 0, stream>>>(qkvw, qw_h, 2304 * 768);
  cast_f32_f16<<<(768 * 768) / 256, 256, 0, stream>>>(oprojw, ow_h, 768 * 768);
  lr_kernel<<<(16384 * 36) / 256, 256, 0, stream>>>(x, lrw, lrb, lr0p, lr1p, lr2p);
  gemm_qkv<<<(1024 * 36) / 8, 256, 0, stream>>>(x_h, qw_h, q_h, k_h, v_h);
  qknorm<<<(96 * 2048) / 256, 256, 0, stream>>>(q_h, k_h);
  dw_kernel<<<96, 256, 0, stream>>>(k_h, v_h, w0, w1, w2, lr0p, lr1p, lr2p, dwb);
  ns5_kernel<<<288, 256, 0, stream>>>(dwb, w0, w1, w2, wnew);
  outhead_kernel<<<96, 256, 0, stream>>>(q_h, wnew, onormw, o_s);
  gemm_oproj<<<(1024 * 12) / 8, 256, 0, stream>>>(o_s, ow_h, out);
}
